// HawkesLSTM_79542794322302
// MI455X (gfx1250) — compile-verified, matrix operand order checked by reading, not run on hardware
//
#include <hip/hip_runtime.h>
#include <hip/hip_bf16.h>

// ---------------- problem constants ----------------
#define T_STEPS 512
#define BATCH   64
#define HID     512
#define PDIM    32
#define KDIM    (PDIM + HID)     // 544
#define NGATE   7
#define NCOL    (NGATE * HID)    // 3584
#define HS      16               // hidden units per workgroup
#define NWG     (HID / HS)       // 32 workgroups
#define THREADS 128              // 4 waves, one 16-batch M-tile each

typedef __attribute__((ext_vector_type(16))) _Float16 v16h;
typedef __attribute__((ext_vector_type(8)))  float    v8f;

struct __align__(16) U4 { unsigned int x, y, z, w; };
struct __align__(16) F4 { float x, y, z, w; };

union Frag { v16h f; U4 u[2]; };

__device__ __forceinline__ float sigmoidf_(float x) {
    return 1.0f / (1.0f + __expf(-x));
}

// ---------------- prep kernels ----------------

// zero h buffer (h0 == 0) and reset barrier counter (deterministic per call)
__global__ void hk_init(_Float16* hbuf, unsigned int* counter) {
    int i = blockIdx.x * blockDim.x + threadIdx.x;
    for (int k = i; k < BATCH * HID; k += gridDim.x * blockDim.x)
        hbuf[k] = (_Float16)0.0f;
    if (i == 0) *counter = 0u;
}

// X[t][b][d] = f16(embed[types[t][b]][d])
__global__ void hk_embed(const float* __restrict__ embed,
                         const int* __restrict__ types,
                         _Float16* __restrict__ X) {
    size_t n = (size_t)T_STEPS * BATCH * PDIM;
    for (size_t i = (size_t)blockIdx.x * blockDim.x + threadIdx.x; i < n;
         i += (size_t)gridDim.x * blockDim.x) {
        size_t tb = i / PDIM;
        int d = (int)(i % PDIM);
        int ty = types[tb];
        X[i] = (_Float16)embed[(size_t)ty * PDIM + d];
    }
}

// Wt[n][k] = f16(W[k][n])  (pre-transpose so B-fragments are contiguous-K rows)
__global__ void hk_wt(const float* __restrict__ W, _Float16* __restrict__ Wt) {
    size_t n = (size_t)NCOL * KDIM;
    for (size_t i = (size_t)blockIdx.x * blockDim.x + threadIdx.x; i < n;
         i += (size_t)gridDim.x * blockDim.x) {
        size_t col = i / KDIM;
        int k = (int)(i % KDIM);
        Wt[i] = (_Float16)W[(size_t)k * NCOL + col];
    }
}

// ---------------- main persistent kernel ----------------
__global__ void __launch_bounds__(THREADS)
hk_main(const float* __restrict__ seq_dt,
        const _Float16* __restrict__ X,
        const _Float16* __restrict__ Wt,
        const float* __restrict__ b_gates,
        _Float16* __restrict__ hbuf,
        unsigned int* counter,
        float* __restrict__ out) {
    extern __shared__ unsigned char smem[];
    _Float16* ldsW = (_Float16*)smem;   // 112 rows x 544 f16 = 119 KB (of 320 KB/WGP)

    const int wg   = blockIdx.x;
    const int j0   = wg * HS;                 // hidden slice base
    const int lane = threadIdx.x & 31;
    const int wave = threadIdx.x >> 5;
    const int m0   = wave * 16;               // batch M-tile base
    const int jl   = lane & 15;               // N index within tile / A-row within tile
    const int half = lane >> 4;
    const int kbase = half * 8;               // A/B K-chunk base per ISA 16-bit layout
    const int b_a  = m0 + jl;                 // batch row this lane loads for A
    const int b_c  = m0 + half * 8;           // C/D batch base (M = b_c + v)

    // ---- stage this WG's weight slice (7 gates x 16 cols x 544 K) into LDS ----
    for (int idx = threadIdx.x; idx < 112 * 68; idx += THREADS) {
        int rl = idx / 68, ch = idx % 68;     // row-local, 16B chunk
        int g = rl >> 4, jj = rl & 15;
        size_t nrow = (size_t)g * HID + j0 + jj;
        U4 v = *(const U4*)(Wt + nrow * KDIM + ch * 8);
        *(U4*)(ldsW + (size_t)rl * KDIM + ch * 8) = v;
    }
    __syncthreads();

    // per-lane gate biases (column-indexed, identical across v)
    float bias[NGATE];
#pragma unroll
    for (int g = 0; g < NGATE; ++g) bias[g] = b_gates[g * HID + j0 + jl];

    // recurrent cell state lives in registers for the whole sequence
    float c[8], ctar[8];
#pragma unroll
    for (int v = 0; v < 8; ++v) { c[v] = 0.0f; ctar[v] = 0.0f; }

    const size_t TBH = (size_t)T_STEPS * BATCH * HID;
    float* out_h  = out;
    float* out_o  = out + TBH;
    float* out_ci = out + 2 * TBH;
    float* out_ct = out + 3 * TBH;
    float* out_dc = out + 4 * TBH;

    for (int t = 0; t < T_STEPS; ++t) {
        v8f acc[NGATE];
#pragma unroll
        for (int g = 0; g < NGATE; ++g)
#pragma unroll
            for (int v = 0; v < 8; ++v) acc[g][v] = 0.0f;

        // ---- k-tile 0: event-embedding part of v = [x, h] (D == 32 exactly) ----
        {
            Frag a;
            const _Float16* ar = X + ((size_t)t * BATCH + b_a) * PDIM;
            a.u[0] = *(const U4*)(ar + kbase);
            a.u[1] = *(const U4*)(ar + kbase + 16);
#pragma unroll
            for (int g = 0; g < NGATE; ++g) {
                Frag b;
                const _Float16* br = ldsW + (size_t)(g * 16 + jl) * KDIM + kbase;
                b.u[0] = *(const U4*)(br);
                b.u[1] = *(const U4*)(br + 16);
                acc[g] = __builtin_amdgcn_wmma_f32_16x16x32_f16(
                    false, a.f, false, b.f, (short)0, acc[g], false, false);
            }
        }
        // ---- k-tiles 1..16: hidden-state part (H == 16*32 exactly) ----
        for (int kt = 1; kt <= 16; ++kt) {
            Frag a;
            const _Float16* ar = hbuf + (size_t)b_a * HID + (kt - 1) * 32;
            a.u[0] = *(const U4*)(ar + kbase);
            a.u[1] = *(const U4*)(ar + kbase + 16);
#pragma unroll
            for (int g = 0; g < NGATE; ++g) {
                Frag b;
                const _Float16* br =
                    ldsW + (size_t)(g * 16 + jl) * KDIM + kt * 32 + kbase;
                b.u[0] = *(const U4*)(br);
                b.u[1] = *(const U4*)(br + 16);
                acc[g] = __builtin_amdgcn_wmma_f32_16x16x32_f16(
                    false, a.f, false, b.f, (short)0, acc[g], false, false);
            }
        }

        // dt for this lane's 8 batch rows (two coalesced 16B loads)
        const F4* dtp = (const F4*)(seq_dt + (size_t)t * BATCH + b_c);
        F4 d0 = dtp[0], d1 = dtp[1];
        float dtv[8] = {d0.x, d0.y, d0.z, d0.w, d1.x, d1.y, d1.z, d1.w};

        // ---- elementwise cell update: all 7 gates are lane/slot-local ----
        size_t obase = (size_t)t * BATCH * HID + (size_t)b_c * HID + (j0 + jl);
#pragma unroll
        for (int v = 0; v < 8; ++v) {
            float gi  = acc[0][v] + bias[0];
            float gf  = acc[1][v] + bias[1];
            float go  = acc[2][v] + bias[2];
            float git = acc[3][v] + bias[3];
            float gft = acc[4][v] + bias[4];
            float gz  = acc[5][v] + bias[5];
            float gd  = acc[6][v] + bias[6];

            float inpt  = sigmoidf_(gi);
            float forget= sigmoidf_(gf);
            float outp  = sigmoidf_(go);
            float intar = sigmoidf_(git);
            float fgtar = sigmoidf_(gft);
            float z     = tanhf(gz);
            float b10   = 10.0f * gd;                       // softplus(beta=10)
            float sp    = (b10 > 20.0f) ? b10 : __logf(1.0f + __expf(b10));
            float decay = sp * 0.1f;

            float ci  = forget * c[v] + inpt * z;
            float ctn = fgtar * ctar[v] + intar * z;
            float ct  = ctn + (ci - ctn) * __expf(-decay * dtv[v]);
            float ht  = outp * tanhf(ct);
            c[v] = ct; ctar[v] = ctn;

            size_t o = obase + (size_t)v * HID;
            out_h[o]  = ht;
            out_o[o]  = outp;
            out_ci[o] = ci;
            out_ct[o] = ctn;
            out_dc[o] = decay;
            hbuf[(size_t)(b_c + v) * HID + (j0 + jl)] = (_Float16)ht;
        }

        // ---- device-wide barrier: publish h, sync all 32 WGs ----
        if (t != T_STEPS - 1) {
            __threadfence();            // release h stores device-wide
            __syncthreads();
            if (threadIdx.x == 0) {
                atomicAdd(counter, 1u);
                unsigned int target = (unsigned)NWG * (unsigned)(t + 1);
                volatile unsigned int* vc = counter;
                while (*vc < target) __builtin_amdgcn_s_sleep(2);
            }
            __syncthreads();
            __threadfence();            // acquire before reading new h
        }
    }
}

// ---------------- launch ----------------
extern "C" void kernel_launch(void* const* d_in, const int* in_sizes, int n_in,
                              void* d_out, int out_size, void* d_ws, size_t ws_size,
                              hipStream_t stream) {
    (void)in_sizes; (void)n_in; (void)out_size; (void)ws_size;
    const float* seq_dt    = (const float*)d_in[0];
    const int*   seq_types = (const int*)d_in[1];
    const float* embed     = (const float*)d_in[2];
    const float* W_gates   = (const float*)d_in[3];
    const float* b_gates   = (const float*)d_in[4];
    // d_in[5..7] = h0/c0/c_target0, all zeros -> handled by hk_init + register init

    unsigned char* ws = (unsigned char*)d_ws;
    _Float16*     Wt      = (_Float16*)(ws);                 // 3584*544*2 = 3,899,392 B
    _Float16*     Xf16    = (_Float16*)(ws + 3899392);       // 512*64*32*2 = 2,097,152 B
    _Float16*     hbuf    = (_Float16*)(ws + 5996544);       // 64*512*2 = 65,536 B
    unsigned int* counter = (unsigned int*)(ws + 6062080);   // 4 B

    hipLaunchKernelGGL(hk_init,  dim3(64),   dim3(256), 0, stream, hbuf, counter);
    hipLaunchKernelGGL(hk_embed, dim3(512),  dim3(256), 0, stream, embed, seq_types, Xf16);
    hipLaunchKernelGGL(hk_wt,    dim3(1024), dim3(256), 0, stream, W_gates, Wt);

    size_t lds = (size_t)112 * KDIM * sizeof(_Float16);      // 121,856 B
    hipFuncSetAttribute((const void*)hk_main,
                        hipFuncAttributeMaxDynamicSharedMemorySize, (int)lds);
    hipLaunchKernelGGL(hk_main, dim3(NWG), dim3(THREADS), lds, stream,
                       seq_dt, Xf16, Wt, b_gates, hbuf, counter, (float*)d_out);
}
